// EncoderBlock_50036368998682
// MI455X (gfx1250) — compile-verified, matrix operand order checked by reading, not run on hardware
//
#include <hip/hip_runtime.h>

// ---------------------------------------------------------------------------
// EncoderBlock for MI455X (gfx1250): bf16 WMMA GEMMs + flash attention.
// Wave32; v_wmma_f32_16x16x32_bf16 for every matrix multiply.
// Double-buffered LDS staging (1 barrier / k-step), packed transposed stores,
// LDS-resident additive mask, global_prefetch for L2 warm-up.
// ---------------------------------------------------------------------------

typedef __bf16 bf16;
typedef bf16  v16bf __attribute__((ext_vector_type(16)));
typedef bf16  v8bf  __attribute__((ext_vector_type(8)));
typedef float v8f   __attribute__((ext_vector_type(8)));

#define B_   2
#define S_   2048
#define D_   1024
#define H_   16
#define DK_  64
#define DFF_ 4096
#define BS_  (B_ * S_)

// Load a 16x32 bf16 A-style fragment from LDS (row-major, leading dim `ld`).
// ISA 16-bit A layout: lane L<16 holds row L, K = 0..7 and 16..23;
// lane L>=16 holds row L-16, K = 8..15 and 24..31.
// B fragments (rows of W / K / V^T) use the identical pattern.
__device__ __forceinline__ v16bf load_frag(const bf16* tile, int ld, int lane) {
    const int row = lane & 15;
    const int kb  = (lane >> 4) << 3;          // 0 or 8
    const bf16* p = tile + row * ld + kb;
    v8bf lo = *(const v8bf*)(p);               // 16B-aligned ds_load_b128
    v8bf hi = *(const v8bf*)(p + 16);
    return __builtin_shufflevector(lo, hi, 0,1,2,3,4,5,6,7,8,9,10,11,12,13,14,15);
}

// ---------------------------------------------------------------------------
// fp32 -> bf16 elementwise cast (weights)
// ---------------------------------------------------------------------------
__global__ void cast_f32_bf16(const float* __restrict__ in, bf16* __restrict__ out, int n) {
    for (int i = blockIdx.x * blockDim.x + threadIdx.x; i < n; i += gridDim.x * blockDim.x)
        out[i] = (bf16)in[i];
}

// ---------------------------------------------------------------------------
// LayerNorm (custom: unbiased std, scalar alpha/beta, /(std+eps)) -> bf16
// ---------------------------------------------------------------------------
__global__ __launch_bounds__(256)
void layernorm_bf16(const float* __restrict__ x, const float* __restrict__ alpha,
                    const float* __restrict__ beta, bf16* __restrict__ out) {
    const int tok = blockIdx.x;
    const int tid = threadIdx.x;
    const float* p = x + (size_t)tok * D_;
    float s = 0.f, s2 = 0.f;
    #pragma unroll
    for (int j = tid; j < D_; j += 256) { float f = p[j]; s += f; s2 += f * f; }
    __shared__ float r1[256], r2[256];
    r1[tid] = s; r2[tid] = s2; __syncthreads();
    for (int st = 128; st > 0; st >>= 1) {
        if (tid < st) { r1[tid] += r1[tid + st]; r2[tid] += r2[tid + st]; }
        __syncthreads();
    }
    const float mean = r1[0] * (1.0f / D_);
    float var = (r2[0] - (float)D_ * mean * mean) * (1.0f / (D_ - 1));
    var = fmaxf(var, 0.f);
    const float inv = alpha[0] / (sqrtf(var) + 1e-6f);
    const float b0  = beta[0];
    #pragma unroll
    for (int j = tid; j < D_; j += 256)
        out[(size_t)tok * D_ + j] = (bf16)((p[j] - mean) * inv + b0);
}

// ---------------------------------------------------------------------------
// GEMM: C[M,N] = A[M,K] @ W[N,K]^T + bias, bf16 in, f32 accum (WMMA).
// EPI 0: bf16 out; EPI 1: bf16 relu out; EPI 2: f32 out = residual + C.
// Tile 128x128 / 256 threads (8 waves, 2x4), wave = 64x32.
// Double-buffered LDS: one barrier per 32-wide k-step.
// ---------------------------------------------------------------------------
template <int EPI>
__global__ __launch_bounds__(256)
void gemm_bf16_wmma(const bf16* __restrict__ A, const bf16* __restrict__ W,
                    const float* __restrict__ bias, const float* __restrict__ res,
                    bf16* __restrict__ outb, float* __restrict__ outf,
                    int M, int N, int K) {
    __shared__ __align__(16) bf16 As[2][128][40];   // +8 pad: 16B-aligned frag reads
    __shared__ __align__(16) bf16 Ws[2][128][40];

    const int tid  = threadIdx.x;
    const int lane = tid & 31;
    const int wave = tid >> 5;
    const int wm   = (wave >> 2) * 64;
    const int wn   = (wave & 3) * 32;
    const int tM   = blockIdx.y * 128;
    const int tN   = blockIdx.x * 128;
    const int lrow = tid >> 1;                   // 0..127
    const int lcol = (tid & 1) * 16;             // 0 or 16

    const bf16* Arow = A + (size_t)(tM + lrow) * K + lcol;
    const bf16* Wrow = W + (size_t)(tN + lrow) * K + lcol;

    v8f acc[4][2];
    #pragma unroll
    for (int i = 0; i < 4; ++i)
        #pragma unroll
        for (int j = 0; j < 2; ++j)
            acc[i][j] = v8f{0.f, 0.f, 0.f, 0.f, 0.f, 0.f, 0.f, 0.f};

    // prologue: stage 0
    {
        const uint4* ag = (const uint4*)(Arow);
        const uint4* wg = (const uint4*)(Wrow);
        uint4 a0 = ag[0], a1 = ag[1], w0 = wg[0], w1 = wg[1];
        *(uint4*)&As[0][lrow][lcol]     = a0;  *(uint4*)&As[0][lrow][lcol + 8] = a1;
        *(uint4*)&Ws[0][lrow][lcol]     = w0;  *(uint4*)&Ws[0][lrow][lcol + 8] = w1;
    }
    __syncthreads();

    const int ksteps = K >> 5;
    int buf = 0;
    for (int s = 0; s < ksteps; ++s) {
        const bool more = (s + 1 < ksteps);
        uint4 a0, a1, w0, w1;
        if (more) {                                   // fetch next stage early
            const int kn = (s + 1) << 5;
            const uint4* ag = (const uint4*)(Arow + kn);
            const uint4* wg = (const uint4*)(Wrow + kn);
            a0 = ag[0]; a1 = ag[1]; w0 = wg[0]; w1 = wg[1];
        }
        if (s + 2 < ksteps) {                         // L2 warm-up two stages ahead
            __builtin_prefetch(Arow + ((s + 2) << 5), 0, 3);
            __builtin_prefetch(Wrow + ((s + 2) << 5), 0, 3);
        }

        v16bf af[4], wf[2];
        #pragma unroll
        for (int i = 0; i < 4; ++i) af[i] = load_frag(&As[buf][wm + i * 16][0], 40, lane);
        #pragma unroll
        for (int j = 0; j < 2; ++j) wf[j] = load_frag(&Ws[buf][wn + j * 16][0], 40, lane);

        #pragma unroll
        for (int i = 0; i < 4; ++i)
            #pragma unroll
            for (int j = 0; j < 2; ++j)
                acc[i][j] = __builtin_amdgcn_wmma_f32_16x16x32_bf16(
                    false, af[i], false, wf[j], (short)0, acc[i][j], false, false);

        if (more) {
            const int nb = buf ^ 1;
            *(uint4*)&As[nb][lrow][lcol]     = a0;  *(uint4*)&As[nb][lrow][lcol + 8] = a1;
            *(uint4*)&Ws[nb][lrow][lcol]     = w0;  *(uint4*)&Ws[nb][lrow][lcol + 8] = w1;
        }
        __syncthreads();
        buf ^= 1;
    }

    // Epilogue. C layout: VGPR e -> row e (lanes 0-15) / e+8 (lanes 16-31).
    const int half = lane >> 4;
    const int lc   = lane & 15;
    #pragma unroll
    for (int j = 0; j < 2; ++j) {
        const int gcol = tN + wn + j * 16 + lc;
        const float bv = bias[gcol];
        #pragma unroll
        for (int i = 0; i < 4; ++i) {
            #pragma unroll
            for (int e = 0; e < 8; ++e) {
                const int grow = tM + wm + i * 16 + e + 8 * half;
                const size_t idx = (size_t)grow * N + gcol;
                const float v = acc[i][j][e] + bv;
                if (EPI == 0)      outb[idx] = (bf16)v;
                else if (EPI == 1) outb[idx] = (bf16)fmaxf(v, 0.f);
                else               outf[idx] = res[idx] + v;
            }
        }
    }
}

// ---------------------------------------------------------------------------
// Flash attention: grid (S/64, H, B), 128 threads (4 waves).
// Double-buffered K / V^T tiles, one barrier per key block.
// ---------------------------------------------------------------------------
__global__ __launch_bounds__(128)
void flash_attn_wmma(const bf16* __restrict__ q, const bf16* __restrict__ k,
                     const bf16* __restrict__ v, const int* __restrict__ mask,
                     bf16* __restrict__ o_out) {
    __shared__ __align__(16) bf16 Qs[64][72];
    __shared__ __align__(16) bf16 Ks[2][64][72];
    __shared__ __align__(16) bf16 Vt[2][64][72];     // V^T: [d][key]
    __shared__ __align__(16) bf16 Ps[4][16][72];     // per-wave P tile
    __shared__ float Madd[S_];                       // additive mask row (0 / -1e9)

    const int qblk = blockIdx.x, h = blockIdx.y, b = blockIdx.z;
    const int tid  = threadIdx.x;
    const int lane = tid & 31;
    const int wave = tid >> 5;
    const int half = lane >> 4;
    const int lc   = lane & 15;
    const int qbase = qblk * 64;

    // thread mappings for staging
    const int rK  = tid >> 1, cK = (tid & 1) * 32;   // K tile: row, 32 cols
    const int k0V = (tid & 31) * 2, c0V = (tid >> 5) * 16; // V: 2 key rows x 16 d cols

    // Load Q block (64 x 64) -> LDS
    {
        const uint4* src = (const uint4*)(q + (size_t)(b * S_ + qbase + rK) * D_ + h * DK_ + cK);
        uint4 d0 = src[0], d1 = src[1], d2 = src[2], d3 = src[3];
        *(uint4*)&Qs[rK][cK]      = d0;
        *(uint4*)&Qs[rK][cK + 8]  = d1;
        *(uint4*)&Qs[rK][cK + 16] = d2;
        *(uint4*)&Qs[rK][cK + 24] = d3;
    }
    // Precompute additive mask row into LDS
    for (int j = tid; j < S_; j += 128)
        Madd[j] = (mask[b * S_ + j] == 0) ? -1e9f : 0.f;

    // Prologue: stage key block 0 into buffer 0
    {
        const uint4* ks = (const uint4*)(k + (size_t)(b * S_ + rK) * D_ + h * DK_ + cK);
        uint4 d0 = ks[0], d1 = ks[1], d2 = ks[2], d3 = ks[3];
        *(uint4*)&Ks[0][rK][cK]      = d0;
        *(uint4*)&Ks[0][rK][cK + 8]  = d1;
        *(uint4*)&Ks[0][rK][cK + 16] = d2;
        *(uint4*)&Ks[0][rK][cK + 24] = d3;
        // V rows k0V, k0V+1, cols c0V..c0V+15 -> packed b32 transposed stores
        __align__(16) unsigned short rA[16], rB[16];
        const uint4* va = (const uint4*)(v + (size_t)(b * S_ + k0V) * D_ + h * DK_ + c0V);
        const uint4* vbp = (const uint4*)(v + (size_t)(b * S_ + k0V + 1) * D_ + h * DK_ + c0V);
        *(uint4*)&rA[0] = va[0];  *(uint4*)&rA[8] = va[1];
        *(uint4*)&rB[0] = vbp[0]; *(uint4*)&rB[8] = vbp[1];
        #pragma unroll
        for (int j = 0; j < 16; ++j)
            *(unsigned int*)&Vt[0][c0V + j][k0V] = (unsigned int)rA[j] | ((unsigned int)rB[j] << 16);
    }
    __syncthreads();

    // Hoisted Q fragments (2 k-steps over DK=64)
    v16bf aq[2];
    #pragma unroll
    for (int ks = 0; ks < 2; ++ks) aq[ks] = load_frag(&Qs[wave * 16][ks * 32], 72, lane);

    float m[8], l[8];
    v8f o[4];
    #pragma unroll
    for (int e = 0; e < 8; ++e) { m[e] = -1e30f; l[e] = 0.f; }
    #pragma unroll
    for (int t = 0; t < 4; ++t) o[t] = v8f{0.f, 0.f, 0.f, 0.f, 0.f, 0.f, 0.f, 0.f};

    const int nblk = S_ / 64;
    int buf = 0;
    for (int kb = 0; kb < nblk; ++kb) {
        const bool more = (kb + 1 < nblk);
        uint4 kreg[4];
        unsigned int pv[16];
        if (more) {     // fetch next K/V block into registers early
            const int base = b * S_ + (kb + 1) * 64;
            const uint4* ks = (const uint4*)(k + (size_t)(base + rK) * D_ + h * DK_ + cK);
            kreg[0] = ks[0]; kreg[1] = ks[1]; kreg[2] = ks[2]; kreg[3] = ks[3];
            __align__(16) unsigned short rA[16], rB[16];
            const uint4* va = (const uint4*)(v + (size_t)(base + k0V) * D_ + h * DK_ + c0V);
            const uint4* vbp = (const uint4*)(v + (size_t)(base + k0V + 1) * D_ + h * DK_ + c0V);
            *(uint4*)&rA[0] = va[0];  *(uint4*)&rA[8] = va[1];
            *(uint4*)&rB[0] = vbp[0]; *(uint4*)&rB[8] = vbp[1];
            #pragma unroll
            for (int j = 0; j < 16; ++j)
                pv[j] = (unsigned int)rA[j] | ((unsigned int)rB[j] << 16);
        }

        // S = Q @ K^T (16 x 64 per wave)
        v8f s4[4];
        #pragma unroll
        for (int t = 0; t < 4; ++t) s4[t] = v8f{0.f, 0.f, 0.f, 0.f, 0.f, 0.f, 0.f, 0.f};
        #pragma unroll
        for (int ks = 0; ks < 2; ++ks) {
            #pragma unroll
            for (int t = 0; t < 4; ++t) {
                v16bf bk = load_frag(&Ks[buf][t * 16][ks * 32], 72, lane);
                s4[t] = __builtin_amdgcn_wmma_f32_16x16x32_bf16(
                    false, aq[ks], false, bk, (short)0, s4[t], false, false);
            }
        }

        // scale + additive mask (from LDS)
        float sc[4][8];
        #pragma unroll
        for (int t = 0; t < 4; ++t) {
            const float madd = Madd[kb * 64 + t * 16 + lc];
            #pragma unroll
            for (int e = 0; e < 8; ++e) sc[t][e] = s4[t][e] * 0.125f + madd;
        }

        // online softmax: row = e + 8*half, reduce over 16-lane N group
        #pragma unroll
        for (int e = 0; e < 8; ++e) {
            float rmax = fmaxf(fmaxf(sc[0][e], sc[1][e]), fmaxf(sc[2][e], sc[3][e]));
            #pragma unroll
            for (int off = 1; off < 16; off <<= 1) rmax = fmaxf(rmax, __shfl_xor(rmax, off, 32));
            const float mnew = fmaxf(m[e], rmax);
            const float alpha = __expf(m[e] - mnew);
            float rsum = 0.f;
            #pragma unroll
            for (int t = 0; t < 4; ++t) { float p = __expf(sc[t][e] - mnew); sc[t][e] = p; rsum += p; }
            #pragma unroll
            for (int off = 1; off < 16; off <<= 1) rsum += __shfl_xor(rsum, off, 32);
            l[e] = l[e] * alpha + rsum;
            m[e] = mnew;
            #pragma unroll
            for (int t = 0; t < 4; ++t) o[t][e] *= alpha;
        }

        // P -> wave-private LDS tile (C layout -> row-major)
        #pragma unroll
        for (int t = 0; t < 4; ++t)
            #pragma unroll
            for (int e = 0; e < 8; ++e)
                Ps[wave][e + 8 * half][t * 16 + lc] = (bf16)sc[t][e];

        // O += P @ V   (B fragment = rows of V^T)
        #pragma unroll
        for (int ks = 0; ks < 2; ++ks) {
            v16bf pa = load_frag(&Ps[wave][0][ks * 32], 72, lane);
            #pragma unroll
            for (int dt = 0; dt < 4; ++dt) {
                v16bf vb2 = load_frag(&Vt[buf][dt * 16][ks * 32], 72, lane);
                o[dt] = __builtin_amdgcn_wmma_f32_16x16x32_bf16(
                    false, pa, false, vb2, (short)0, o[dt], false, false);
            }
        }

        // stage next block into the other buffer
        if (more) {
            const int nb = buf ^ 1;
            *(uint4*)&Ks[nb][rK][cK]      = kreg[0];
            *(uint4*)&Ks[nb][rK][cK + 8]  = kreg[1];
            *(uint4*)&Ks[nb][rK][cK + 16] = kreg[2];
            *(uint4*)&Ks[nb][rK][cK + 24] = kreg[3];
            #pragma unroll
            for (int j = 0; j < 16; ++j)
                *(unsigned int*)&Vt[nb][c0V + j][k0V] = pv[j];
        }
        __syncthreads();
        buf ^= 1;
    }

    // normalize + write bf16 [BS, D]
    #pragma unroll
    for (int dt = 0; dt < 4; ++dt) {
        #pragma unroll
        for (int e = 0; e < 8; ++e) {
            const int row = qbase + wave * 16 + e + 8 * half;
            const int col = h * DK_ + dt * 16 + lc;
            const float val = o[dt][e] / l[e];
            o_out[(size_t)(b * S_ + row) * D_ + col] = (bf16)val;
        }
    }
}

// ---------------------------------------------------------------------------
// Host-side orchestration
// ---------------------------------------------------------------------------
extern "C" void kernel_launch(void* const* d_in, const int* in_sizes, int n_in,
                              void* d_out, int out_size, void* d_ws, size_t ws_size,
                              hipStream_t stream) {
    const float* x   = (const float*)d_in[0];
    const int*   msk = (const int*)d_in[1];
    const float* wq  = (const float*)d_in[2];   const float* bq  = (const float*)d_in[3];
    const float* wk  = (const float*)d_in[4];   const float* bk  = (const float*)d_in[5];
    const float* wv  = (const float*)d_in[6];   const float* bv  = (const float*)d_in[7];
    const float* wo  = (const float*)d_in[8];   const float* bo  = (const float*)d_in[9];
    const float* w1  = (const float*)d_in[10];  const float* b1  = (const float*)d_in[11];
    const float* w2  = (const float*)d_in[12];  const float* b2  = (const float*)d_in[13];
    const float* a1  = (const float*)d_in[14];  const float* be1 = (const float*)d_in[15];
    const float* a2  = (const float*)d_in[16];  const float* be2 = (const float*)d_in[17];
    float* out = (float*)d_out;

    char* ws = (char*)d_ws;
    size_t off = 0;
    auto carve = [&](size_t bytes) { char* p = ws + off; off += (bytes + 255) & ~(size_t)255; return p; };
    bf16*  wqb  = (bf16*)carve((size_t)D_ * D_ * 2);
    bf16*  wkb  = (bf16*)carve((size_t)D_ * D_ * 2);
    bf16*  wvb  = (bf16*)carve((size_t)D_ * D_ * 2);
    bf16*  wob  = (bf16*)carve((size_t)D_ * D_ * 2);
    bf16*  w1b  = (bf16*)carve((size_t)DFF_ * D_ * 2);
    bf16*  w2b  = (bf16*)carve((size_t)D_ * DFF_ * 2);
    bf16*  xn   = (bf16*)carve((size_t)BS_ * D_ * 2);
    bf16*  qb   = (bf16*)carve((size_t)BS_ * D_ * 2);
    bf16*  kb   = (bf16*)carve((size_t)BS_ * D_ * 2);
    bf16*  vb   = (bf16*)carve((size_t)BS_ * D_ * 2);
    bf16*  atb  = (bf16*)carve((size_t)BS_ * D_ * 2);
    float* x1   = (float*)carve((size_t)BS_ * D_ * 4);
    bf16*  hb   = (bf16*)carve((size_t)BS_ * DFF_ * 2);

    // 1) weights -> bf16
    cast_f32_bf16<<<2048, 256, 0, stream>>>(wq, wqb, D_ * D_);
    cast_f32_bf16<<<2048, 256, 0, stream>>>(wk, wkb, D_ * D_);
    cast_f32_bf16<<<2048, 256, 0, stream>>>(wv, wvb, D_ * D_);
    cast_f32_bf16<<<2048, 256, 0, stream>>>(wo, wob, D_ * D_);
    cast_f32_bf16<<<4096, 256, 0, stream>>>(w1, w1b, DFF_ * D_);
    cast_f32_bf16<<<4096, 256, 0, stream>>>(w2, w2b, D_ * DFF_);

    // 2) LN1
    layernorm_bf16<<<BS_, 256, 0, stream>>>(x, a1, be1, xn);

    // 3) Q,K,V projections
    dim3 gD(D_ / 128, BS_ / 128);
    gemm_bf16_wmma<0><<<gD, 256, 0, stream>>>(xn, wqb, bq, nullptr, qb, nullptr, BS_, D_, D_);
    gemm_bf16_wmma<0><<<gD, 256, 0, stream>>>(xn, wkb, bk, nullptr, kb, nullptr, BS_, D_, D_);
    gemm_bf16_wmma<0><<<gD, 256, 0, stream>>>(xn, wvb, bv, nullptr, vb, nullptr, BS_, D_, D_);

    // 4) attention
    flash_attn_wmma<<<dim3(S_ / 64, H_, B_), 128, 0, stream>>>(qb, kb, vb, msk, atb);

    // 5) out-proj + residual -> x1 (f32)
    gemm_bf16_wmma<2><<<gD, 256, 0, stream>>>(atb, wob, bo, x, nullptr, x1, BS_, D_, D_);

    // 6) LN2
    layernorm_bf16<<<BS_, 256, 0, stream>>>(x1, a2, be2, xn);

    // 7) FFN up + ReLU (bf16)
    dim3 gF(DFF_ / 128, BS_ / 128);
    gemm_bf16_wmma<1><<<gF, 256, 0, stream>>>(xn, w1b, b1, nullptr, hb, nullptr, BS_, DFF_, D_);

    // 8) FFN down + residual -> out (f32)
    gemm_bf16_wmma<2><<<gD, 256, 0, stream>>>(hb, w2b, b2, x1, nullptr, out, BS_, D_, DFF_);

    (void)in_sizes; (void)n_in; (void)out_size; (void)ws_size;
}
